// OverlapGATNet_18193481466143
// MI455X (gfx1250) — matrix operand order repaired and checked by reading, not compiled, not run on hardware
//
#include <hip/hip_runtime.h>
#include <math.h>

// ---------------------------------------------------------------------------
// Types for CDNA5 WMMA (wave32): v16h A/B fragments, v8f C/D accumulators.
// ---------------------------------------------------------------------------
typedef _Float16 h16;
typedef _Float16 v16h  __attribute__((ext_vector_type(16)));
typedef _Float16 half8 __attribute__((ext_vector_type(8)));
typedef float    v8f   __attribute__((ext_vector_type(8)));
typedef unsigned int uint4v __attribute__((ext_vector_type(4)));
typedef int      int4v  __attribute__((ext_vector_type(4)));
typedef int      int8v  __attribute__((ext_vector_type(8)));

#define NB      8
#define GHh     8
#define GWw     112
#define NPATCH  896
#define NP      (NB*NPATCH)          // 7168 patches
#define TOPK    300
#define NREL    121                  // (2*5+1)^2
#define E_NEIGH (TOPK*NREL)          // 36300
#define E_TOT   (E_NEIGH + NPATCH)   // 37196
#define EPSBN   1e-5f
#define KMAXLDS 1152                 // largest staged K (res3): 16*1152*2B = 36 KB

__device__ __forceinline__ v8f wmma16(v16h a, v16h b, v8f c) {
  // emits v_wmma_f32_16x16x32_f16
  return __builtin_amdgcn_wmma_f32_16x16x32_f16(false, a, false, b, (short)0, c,
                                                false, false);
}

#if __has_builtin(__builtin_amdgcn_tensor_load_to_lds) && \
    __has_builtin(__builtin_amdgcn_s_wait_tensorcnt)
#define HAVE_TDM 1
#else
#define HAVE_TDM 0
#endif

// ---------------------------------------------------------------------------
// Stage a 16-row x K-element f16 weight tile (rows contiguous, row stride = K)
// from global into LDS.  On CDNA5 this uses the Tensor Data Mover: one wave
// issues tensor_load_to_lds with a 2-D D# descriptor, waits on TENSORcnt,
// then the workgroup barrier publishes the tile.  Fallback: cooperative copy.
// `lds` must be the first (offset-0) static LDS object of the kernel.
// ---------------------------------------------------------------------------
__device__ __forceinline__ void stage_btile(const h16* __restrict__ g,
                                            h16* __restrict__ lds, int K, int Nrows) {
#if HAVE_TDM
  if (threadIdx.x < 32) {            // wave 0 issues the DMA (TDM ignores EXEC)
    unsigned long long ga = (unsigned long long)(const void*)g;
    uint4v g0 = {};
    g0[0] = 1u;                                      // count=1, user descriptor
    g0[1] = 0u;                                      // lds_addr: stage buf at LDS offset 0
    g0[2] = (unsigned)ga;                            // global_addr[31:0]
    g0[3] = (unsigned)((ga >> 32) & 0x01FFFFFFull)   // global_addr[56:32]
            | 0x80000000u;                           // type=2 ("image")
    int8v g1 = {};
    g1[0] = 1 << 16;                                 // data_size=1 (2 bytes)
    g1[1] = (K & 0xffff) << 16;                      // tensor_dim0[15:0]
    g1[2] = ((K >> 16) & 0xffff)                     // tensor_dim0[31:16]
            | ((Nrows & 0xffff) << 16);              // tensor_dim1[15:0]
    g1[3] = ((Nrows >> 16) & 0xffff)                 // tensor_dim1[31:16]
            | ((K & 0xffff) << 16);                  // tile_dim0 = K
    g1[4] = 16;                                      // tile_dim1 = 16 rows
    g1[5] = K;                                       // tensor_dim0_stride[31:0]
    g1[6] = 0; g1[7] = 0;
    int4v z = {};
#if defined(__clang_major__) && __clang_major__ >= 23
    int8v z8 = {};
    __builtin_amdgcn_tensor_load_to_lds(g0, g1, z, z, z8, 0);
#else
    __builtin_amdgcn_tensor_load_to_lds(g0, g1, z, z, 0);
#endif
    __builtin_amdgcn_s_wait_tensorcnt(0);
  }
#else
  {
    const half8* gs = (const half8*)g;
    half8* ls = (half8*)lds;
    int n8 = (16 * K) >> 3;
    for (int i = threadIdx.x; i < n8; i += blockDim.x) ls[i] = gs[i];
  }
#endif
  __syncthreads();
}

// ordered-uint encoding of float for atomicMax-based segment_max
__device__ __forceinline__ unsigned fenc(float f) {
  unsigned u = __float_as_uint(f);
  return (u & 0x80000000u) ? ~u : (u | 0x80000000u);
}
__device__ __forceinline__ float fdec(unsigned u) {
  u = (u & 0x80000000u) ? (u & 0x7fffffffu) : ~u;
  return __uint_as_float(u);
}

// ---------------------------------------------------------------------------
// Weight / activation packing
// ---------------------------------------------------------------------------
__global__ void pack_conv_w(const float* __restrict__ w, h16* __restrict__ wt,
                            int Cout, int Cin, int Kpad) {
  int idx = blockIdx.x * blockDim.x + threadIdx.x;
  if (idx >= Cout * Kpad) return;
  int co = idx / Kpad, k = idx - co * Kpad;
  float v = 0.f;
  if (k < 9 * Cin) {                 // K order: tap-major, cin-minor
    int tap = k / Cin, cin = k - tap * Cin;
    v = w[(co * Cin + cin) * 9 + tap];   // OIHW
  }
  wt[idx] = (h16)v;
}

// dense [K][N] f32 -> [N][K] f16 (K-contiguous rows for B-fragment loads)
__global__ void pack_mat(const float* __restrict__ w, h16* __restrict__ bt,
                         int K, int N) {
  long long idx = blockIdx.x * (long long)blockDim.x + threadIdx.x;
  if (idx >= (long long)K * N) return;
  int n = (int)(idx / K), k = (int)(idx - (long long)n * K);
  bt[idx] = (h16)w[(long long)k * N + n];
}

__global__ void patchify(const float* __restrict__ x, h16* __restrict__ out) {
  int idx = blockIdx.x * blockDim.x + threadIdx.x;
  if (idx >= NP * 64) return;
  int p = idx >> 6, q = idx & 63;
  int ph = q >> 3, pw = q & 7;
  int b = p / NPATCH, n = p - b * NPATCH;
  int gh = n / GWw, gw = n - gh * GWw;
  out[idx] = (h16)x[((b * 64) + gh * 8 + ph) * 900 + gw * 8 + pw];
}

// ---------------------------------------------------------------------------
// Implicit-GEMM 3x3 conv (pad 1), NHWC f16 activations, WMMA 16x16x32.
// B tile (16 x Kpad weights) is TDM-staged into LDS once per workgroup.
// Epilogue: BN-fold (scale*rsqrt(1+eps), bias), optional residual, ReLU.
// One wave computes one 16x16 output tile. K order = tap*Cin + cin.
// ---------------------------------------------------------------------------
__global__ void conv_wmma(const h16* __restrict__ act, const h16* __restrict__ wt,
                          const float* __restrict__ scale, const float* __restrict__ bias,
                          const h16* __restrict__ resid, h16* __restrict__ outp,
                          int NPt, int IH, int IW, int Cin,
                          int OH, int OW, int Cout, int Kpad, int stride, int relu) {
  __shared__ __align__(16) h16 bstage[16 * KMAXLDS];
  const int M = NPt * OH * OW;
  int wave = threadIdx.x >> 5, lane = threadIdx.x & 31;
  int mtile = blockIdx.x * 4 + wave;
  int col0 = blockIdx.y * 16;
  stage_btile(wt + (size_t)col0 * Kpad, bstage, Kpad, Cout);  // all threads
  if (mtile * 16 >= M) return;            // wave-uniform exit, after barrier
  int l15 = lane & 15, hi = lane >> 4;
  int m = mtile * 16 + l15;
  int p = m / (OH * OW);
  int rem = m - p * (OH * OW);
  int oy = rem / OW, ox = rem - oy * OW;
  const float rsq = rsqrtf(1.0f + EPSBN);
  v8f acc = {};
  const int ksteps = Kpad >> 5;
  const int vec = (Cin & 7) == 0;
  for (int t = 0; t < ksteps; ++t) {
    int k0 = t << 5;
    v16h afrag, bfrag;
    // --- A fragment: lane<16 holds K=k0..k0+15; lane>=16: k0+16..k0+31
    #pragma unroll
    for (int g = 0; g < 2; ++g) {
      int kb = k0 + hi * 16 + g * 8;
      if (vec) {                            // Cin%8==0: whole group = one tap
        int tap = kb / Cin, cin0 = kb - tap * Cin;
        int dy = tap / 3, dx = tap - dy * 3;
        int iy = oy * stride + dy - 1, ix = ox * stride + dx - 1;
        half8 v = {};
        if (iy >= 0 && iy < IH && ix >= 0 && ix < IW && tap < 9)
          v = *(const half8*)(act + (size_t)(((p * IH + iy) * IW + ix) * Cin + cin0));
        #pragma unroll
        for (int j = 0; j < 8; ++j) afrag[g * 8 + j] = v[j];
      } else {                              // conv1 (Cin=1, K padded 9->32)
        #pragma unroll
        for (int j = 0; j < 8; ++j) {
          int k = kb + j;
          h16 v = (h16)0.f;
          if (k < 9 * Cin) {
            int tap = k / Cin, cin = k - tap * Cin;
            int dy = tap / 3, dx = tap - dy * 3;
            int iy = oy * stride + dy - 1, ix = ox * stride + dx - 1;
            if (iy >= 0 && iy < IH && ix >= 0 && ix < IW)
              v = act[((p * IH + iy) * IW + ix) * Cin + cin];
          }
          afrag[g * 8 + j] = v;
        }
      }
    }
    // --- B fragment from LDS: lane n holds col n, K = k0+hi*16 .. +15
    {
      const h16* bp = bstage + (size_t)l15 * Kpad + k0 + hi * 16;
      half8 b0 = *(const half8*)bp;
      half8 b1 = *(const half8*)(bp + 8);
      #pragma unroll
      for (int j = 0; j < 8; ++j) { bfrag[j] = b0[j]; bfrag[j + 8] = b1[j]; }
    }
    acc = wmma16(afrag, bfrag, acc);
  }
  // --- epilogue (C layout: vgpr r -> M = r + 8*hi, N = lane&15)
  int n = col0 + l15;
  float s = scale ? scale[n] * rsq : 1.f;
  float bv = bias ? bias[n] : 0.f;
  #pragma unroll
  for (int r = 0; r < 8; ++r) {
    int mr = mtile * 16 + r + hi * 8;
    float v = acc[r] * s + bv;
    if (resid) v += (float)resid[(size_t)mr * Cout + n];
    if (relu) v = fmaxf(v, 0.f);
    outp[(size_t)mr * Cout + n] = (h16)v;
  }
}

// ---------------------------------------------------------------------------
// Generic WMMA GEMM: out = act( (A[M][K] * Bt[N][K]^T) * scale + bias )
// act: 0=none 1=relu 2=sigmoid.  scale (if set) is BN-folded with rsqrt(1+eps).
// K <= 1152: B tile staged to LDS via TDM; otherwise direct global stream.
// ---------------------------------------------------------------------------
__global__ void gemm_wmma(const h16* __restrict__ A, const h16* __restrict__ Bt,
                          const float* __restrict__ scale, const float* __restrict__ bias,
                          float* __restrict__ outF, h16* __restrict__ outH,
                          int M, int N, int K, int actf) {
  __shared__ __align__(16) h16 bstage[16 * KMAXLDS];
  int wave = threadIdx.x >> 5, lane = threadIdx.x & 31;
  int mtile = blockIdx.x * 4 + wave;
  int col0 = blockIdx.y * 16;
  const bool useLds = (K <= KMAXLDS);             // block-uniform
  if (useLds) stage_btile(Bt + (size_t)col0 * K, bstage, K, N);
  if (mtile >= (M + 15) / 16) return;             // after barrier
  int l15 = lane & 15, hi = lane >> 4;
  int row = mtile * 16 + l15;
  const float rsq = rsqrtf(1.0f + EPSBN);
  v8f acc = {};
  for (int k0 = 0; k0 < K; k0 += 32) {
    v16h afrag = {}, bfrag;
    if (row < M) {
      const h16* ap = A + (size_t)row * K + k0 + hi * 16;
      half8 a0 = *(const half8*)ap;
      half8 a1 = *(const half8*)(ap + 8);
      #pragma unroll
      for (int j = 0; j < 8; ++j) { afrag[j] = a0[j]; afrag[j + 8] = a1[j]; }
    }
    if (useLds) {                                  // ds_load path
      const h16* bp = bstage + (size_t)l15 * K + k0 + hi * 16;
      half8 b0 = *(const half8*)bp;
      half8 b1 = *(const half8*)(bp + 8);
      #pragma unroll
      for (int j = 0; j < 8; ++j) { bfrag[j] = b0[j]; bfrag[j + 8] = b1[j]; }
    } else {                                       // global stream (K=16384)
      const h16* bp = Bt + (size_t)(col0 + l15) * K + k0 + hi * 16;
      half8 b0 = *(const half8*)bp;
      half8 b1 = *(const half8*)(bp + 8);
      #pragma unroll
      for (int j = 0; j < 8; ++j) { bfrag[j] = b0[j]; bfrag[j + 8] = b1[j]; }
      if (k0 + 32 < K) __builtin_prefetch(bp + 32, 0, 0);
    }
    acc = wmma16(afrag, bfrag, acc);
  }
  int n = col0 + l15;
  float s = scale ? scale[n] * rsq : 1.f;
  float bv = bias ? bias[n] : 0.f;
  #pragma unroll
  for (int r = 0; r < 8; ++r) {
    int mr = mtile * 16 + r + hi * 8;
    if (mr >= M) continue;
    float v = acc[r] * s + bv;
    if (actf == 1) v = fmaxf(v, 0.f);
    else if (actf == 2) v = 1.f / (1.f + expf(-v));
    if (outF) outF[(size_t)mr * N + n] = v;
    if (outH) outH[(size_t)mr * N + n] = (h16)v;
  }
}

// ---------------------------------------------------------------------------
// Mean pool 2x2 -> feat (f32 + f16)
// ---------------------------------------------------------------------------
__global__ void meanpool_feat(const h16* __restrict__ act,
                              float* __restrict__ fF, h16* __restrict__ fH) {
  int idx = blockIdx.x * blockDim.x + threadIdx.x;
  if (idx >= NP * 128) return;
  int p = idx >> 7, c = idx & 127;
  float s = 0.f;
  for (int q = 0; q < 4; ++q) s += (float)act[(size_t)(p * 4 + q) * 128 + c];
  s *= 0.25f;
  fF[idx] = s; fH[idx] = (h16)s;
}

// ---------------------------------------------------------------------------
// Per-graph top-K by L2 score (iterative argmax in LDS, ties -> lowest index)
// ---------------------------------------------------------------------------
__global__ void topk_kernel(const float* __restrict__ featF, int* __restrict__ topk) {
  __shared__ float sc[NPATCH];
  __shared__ float rv[256];
  __shared__ int   ri[256];
  int b = blockIdx.x, tid = threadIdx.x;
  for (int n = tid; n < NPATCH; n += 256) {
    float s = 0.f;
    const float* f = featF + (size_t)(b * NPATCH + n) * 128;
    for (int c = 0; c < 128; ++c) { float v = f[c]; s += v * v; }
    sc[n] = sqrtf(s);
  }
  __syncthreads();
  for (int it = 0; it < TOPK; ++it) {
    float best = -1e30f; int bi = 0x7fffffff;
    for (int n = tid; n < NPATCH; n += 256) {
      float v = sc[n];
      if (v > best || (v == best && n < bi)) { best = v; bi = n; }
    }
    rv[tid] = best; ri[tid] = bi; __syncthreads();
    for (int off = 128; off > 0; off >>= 1) {
      if (tid < off) {
        float v = rv[tid + off]; int i = ri[tid + off];
        if (v > rv[tid] || (v == rv[tid] && i < ri[tid])) { rv[tid] = v; ri[tid] = i; }
      }
      __syncthreads();
    }
    if (tid == 0) { topk[b * TOPK + it] = ri[0]; sc[ri[0]] = -1e30f; }
    __syncthreads();
  }
}

__global__ void build_edges(const int* __restrict__ topk, int* __restrict__ src,
                            int* __restrict__ dst, int* __restrict__ mask) {
  int idx = blockIdx.x * blockDim.x + threadIdx.x;
  if (idx >= NB * E_TOT) return;
  int b = idx / E_TOT, e = idx - b * E_TOT;
  int s, d, mk;
  if (e < E_NEIGH) {
    int i = e / NREL, r = e - i * NREL;
    int node = topk[b * TOPK + i];
    int y = node / GWw, x = node - y * GWw;
    int yn = y + r / 11 - 5, xn = x + r % 11 - 5;
    int valid = (yn >= 0 && yn < GHh && xn >= 0 && xn < GWw);
    s = node; d = valid ? yn * GWw + xn : 0; mk = valid;
  } else { int n = e - E_NEIGH; s = n; d = n; mk = 1; }
  src[idx] = s; dst[idx] = d; mask[idx] = mk;
}

__global__ void init_seg(unsigned* __restrict__ emax, float* __restrict__ denom, int n) {
  int i = blockIdx.x * blockDim.x + threadIdx.x;
  if (i < n) { emax[i] = fenc(-1e30f); denom[i] = 0.f; }
}

__global__ void edge_score(const float* __restrict__ xl, const float* __restrict__ xr,
                           const float* __restrict__ att, const int* __restrict__ src,
                           const int* __restrict__ dst, const int* __restrict__ mask,
                           float* __restrict__ esc, unsigned* __restrict__ emax, int dout) {
  int idx = blockIdx.x * blockDim.x + threadIdx.x;
  if (idx >= NB * E_TOT) return;
  int b = idx / E_TOT;
  int sg = b * NPATCH + src[idx], dg = b * NPATCH + dst[idx];
  float e = -1e9f;
  if (mask[idx]) {
    const float* pl = xl + (size_t)sg * dout;
    const float* pr = xr + (size_t)dg * dout;
    float sum = 0.f;
    for (int j = 0; j < dout; ++j) {
      float v = pl[j] + pr[j];
      v = v > 0.f ? v : 0.2f * v;            // leaky_relu
      sum += v * att[j];
    }
    e = sum;
    atomicMax(emax + dg, fenc(e));
  }
  esc[idx] = e;
}

__global__ void edge_exp(const float* __restrict__ esc, const int* __restrict__ dst,
                         const int* __restrict__ mask, const unsigned* __restrict__ emax,
                         float* __restrict__ ex, float* __restrict__ denom) {
  int idx = blockIdx.x * blockDim.x + threadIdx.x;
  if (idx >= NB * E_TOT) return;
  int b = idx / E_TOT, dg = b * NPATCH + dst[idx];
  float v = 0.f;
  if (mask[idx]) { v = expf(esc[idx] - fdec(emax[dg])); atomicAdd(denom + dg, v); }
  ex[idx] = v;
}

__global__ void edge_aggr(const float* __restrict__ ex, const float* __restrict__ denom,
                          const float* __restrict__ xl, const int* __restrict__ src,
                          const int* __restrict__ dst, const int* __restrict__ mask,
                          float* __restrict__ outp, int dout) {
  long long total = (long long)NB * E_TOT * dout;
  for (long long t = blockIdx.x * (long long)blockDim.x + threadIdx.x; t < total;
       t += (long long)gridDim.x * blockDim.x) {
    int j = (int)(t % dout);
    long long e = t / dout;
    if (!mask[e]) continue;
    int b = (int)(e / E_TOT);
    int sg = b * NPATCH + src[e], dg = b * NPATCH + dst[e];
    float alpha = ex[e] / denom[dg];
    atomicAdd(outp + (size_t)dg * dout + j, alpha * xl[(size_t)sg * dout + j]);
  }
}

__global__ void relu_store(float* __restrict__ io, h16* __restrict__ oh, long long n) {
  long long i = blockIdx.x * (long long)blockDim.x + threadIdx.x;
  if (i >= n) return;
  float v = fmaxf(io[i], 0.f);
  io[i] = v;
  if (oh) oh[i] = (h16)v;
}

// gather h2[topk] and L2-normalize (axis=feature, 256) -> g
__global__ void gather_norm(const float* __restrict__ h2, const int* __restrict__ topk,
                            float* __restrict__ gF, h16* __restrict__ gH) {
  __shared__ float red[256];
  int bi = blockIdx.x, d = threadIdx.x;
  int b = bi / TOPK, i = bi - b * TOPK;
  int node = topk[b * TOPK + i];
  float v = h2[(size_t)(b * NPATCH + node) * 256 + d];
  red[d] = v * v; __syncthreads();
  for (int off = 128; off > 0; off >>= 1) { if (d < off) red[d] += red[d + off]; __syncthreads(); }
  float nrm = fmaxf(sqrtf(red[0]), 1e-12f);
  float o = v / nrm;
  gF[(size_t)bi * 256 + d] = o; gH[(size_t)bi * 256 + d] = (h16)o;
}

__global__ void softmax64(float* __restrict__ a) {    // [rows][64] in place
  __shared__ float red[64];
  int r = blockIdx.x, c = threadIdx.x;
  float v = a[(size_t)r * 64 + c];
  red[c] = v; __syncthreads();
  for (int off = 32; off > 0; off >>= 1) { if (c < off) red[c] = fmaxf(red[c], red[c + off]); __syncthreads(); }
  float m = red[0]; __syncthreads();
  float e = expf(v - m);
  red[c] = e; __syncthreads();
  for (int off = 32; off > 0; off >>= 1) { if (c < off) red[c] += red[c + off]; __syncthreads(); }
  a[(size_t)r * 64 + c] = e / red[0];
}

__global__ void act_sum(const float* __restrict__ a, float* __restrict__ s) {
  int idx = blockIdx.x * blockDim.x + threadIdx.x;
  if (idx >= NB * 64) return;
  int b = idx / 64, c = idx & 63;
  float t = 0.f;
  for (int k = 0; k < TOPK; ++k) t += a[((size_t)b * TOPK + k) * 64 + c];
  s[idx] = t;
}

__global__ void vlad_kernel(const float* __restrict__ act, const float* __restrict__ gF,
                            const float* __restrict__ cw2, const float* __restrict__ asum,
                            float* __restrict__ vlad) {
  int idx = blockIdx.x * blockDim.x + threadIdx.x;
  if (idx >= NB * 16384) return;
  int b = idx / 16384, rem = idx - b * 16384;
  int d = rem >> 6, c = rem & 63;
  float t = 0.f;
  for (int k = 0; k < TOPK; ++k)
    t += act[((size_t)b * TOPK + k) * 64 + c] * gF[((size_t)b * TOPK + k) * 256 + d];
  t -= asum[b * 64 + c] * cw2[d * 64 + c];
  vlad[idx] = t;
}

__global__ void vlad_intra_norm(float* __restrict__ vlad) { // norm over D per (b,c)
  __shared__ float red[256];
  int bc = blockIdx.x, b = bc >> 6, c = bc & 63, d = threadIdx.x;
  float v = vlad[(size_t)b * 16384 + d * 64 + c];
  red[d] = v * v; __syncthreads();
  for (int off = 128; off > 0; off >>= 1) { if (d < off) red[d] += red[d + off]; __syncthreads(); }
  float nrm = fmaxf(sqrtf(red[0]), 1e-12f);
  vlad[(size_t)b * 16384 + d * 64 + c] = v / nrm;
}

__global__ void vlad_global_norm(float* __restrict__ vlad, h16* __restrict__ vh) {
  __shared__ float red[256];
  int b = blockIdx.x, tid = threadIdx.x;
  float s = 0.f;
  for (int i = tid; i < 16384; i += 256) { float v = vlad[(size_t)b * 16384 + i]; s += v * v; }
  red[tid] = s; __syncthreads();
  for (int off = 128; off > 0; off >>= 1) { if (tid < off) red[tid] += red[tid + off]; __syncthreads(); }
  float nrm = fmaxf(sqrtf(red[0]), 1e-12f);
  for (int i = tid; i < 16384; i += 256) {
    float v = vlad[(size_t)b * 16384 + i] / nrm;
    vlad[(size_t)b * 16384 + i] = v;
    vh[(size_t)b * 16384 + i] = (h16)v;
  }
}

__global__ void final_out(const float* __restrict__ hid, const float* __restrict__ gates,
                          float* __restrict__ outp) {
  __shared__ float red[256];
  int b = blockIdx.x, tid = threadIdx.x;
  float v = hid[b * 256 + tid] * gates[b * 256 + tid];
  red[tid] = v * v; __syncthreads();
  for (int off = 128; off > 0; off >>= 1) { if (tid < off) red[tid] += red[tid + off]; __syncthreads(); }
  float nrm = fmaxf(sqrtf(red[0]), 1e-12f);
  outp[b * 256 + tid] = v / nrm;
}

// ---------------------------------------------------------------------------
// Orchestration
// ---------------------------------------------------------------------------
extern "C" void kernel_launch(void* const* d_in, const int* in_sizes, int n_in,
                              void* d_out, int out_size, void* d_ws, size_t ws_size,
                              hipStream_t stream) {
  (void)in_sizes; (void)n_in; (void)out_size; (void)ws_size;
  // --- input mapping (recursive insertion-order flatten of setup_inputs) ---
  const float* x      = (const float*)d_in[0];
  const float* c1w    = (const float*)d_in[1];
  const float* bn1g   = (const float*)d_in[2];
  const float* bn1b   = (const float*)d_in[3];
  const float* r1w1   = (const float*)d_in[4];
  const float* r1g1   = (const float*)d_in[5];
  const float* r1b1   = (const float*)d_in[6];
  const float* r1w2   = (const float*)d_in[7];
  const float* r1g2   = (const float*)d_in[8];
  const float* r1b2   = (const float*)d_in[9];
  const float* d1w    = (const float*)d_in[10];
  const float* bnd1g  = (const float*)d_in[11];
  const float* bnd1b  = (const float*)d_in[12];
  const float* r2w1   = (const float*)d_in[13];
  const float* r2g1   = (const float*)d_in[14];
  const float* r2b1   = (const float*)d_in[15];
  const float* r2w2   = (const float*)d_in[16];
  const float* r2g2   = (const float*)d_in[17];
  const float* r2b2   = (const float*)d_in[18];
  const float* d2w    = (const float*)d_in[19];
  const float* bnd2g  = (const float*)d_in[20];
  const float* bnd2b  = (const float*)d_in[21];
  const float* r3w1   = (const float*)d_in[22];
  const float* r3g1   = (const float*)d_in[23];
  const float* r3b1   = (const float*)d_in[24];
  const float* r3w2   = (const float*)d_in[25];
  const float* r3g2   = (const float*)d_in[26];
  const float* r3b2   = (const float*)d_in[27];
  const float* g1wl   = (const float*)d_in[28];
  const float* g1bl   = (const float*)d_in[29];
  const float* g1wr   = (const float*)d_in[30];
  const float* g1br   = (const float*)d_in[31];
  const float* g1att  = (const float*)d_in[32];
  const float* g1wres = (const float*)d_in[33];
  const float* g1bias = (const float*)d_in[34];
  const float* g2wl   = (const float*)d_in[35];
  const float* g2bl   = (const float*)d_in[36];
  const float* g2wr   = (const float*)d_in[37];
  const float* g2br   = (const float*)d_in[38];
  const float* g2att  = (const float*)d_in[39];
  const float* g2wres = (const float*)d_in[40];
  const float* g2bias = (const float*)d_in[41];
  const float* vcw    = (const float*)d_in[42];
  const float* vcb    = (const float*)d_in[43];
  const float* vcw2   = (const float*)d_in[44];
  const float* vhw    = (const float*)d_in[45];
  const float* vbn2g  = (const float*)d_in[46];
  const float* vbn2b  = (const float*)d_in[47];
  const float* vgw    = (const float*)d_in[48];
  const float* vgb    = (const float*)d_in[49];

  // --- workspace carving ---
  size_t off = 0;
  auto alloc = [&](size_t bytes) -> void* {
    void* p = (char*)d_ws + off;
    off += (bytes + 255) & ~(size_t)255;
    return p;
  };
  h16* patchIn = (h16*)alloc((size_t)NP * 64 * 2);
  h16* wtC1  = (h16*)alloc(32 * 32 * 2);
  h16* wtR1a = (h16*)alloc(32 * 288 * 2);
  h16* wtR1b = (h16*)alloc(32 * 288 * 2);
  h16* wtD1  = (h16*)alloc(64 * 288 * 2);
  h16* wtR2a = (h16*)alloc(64 * 576 * 2);
  h16* wtR2b = (h16*)alloc(64 * 576 * 2);
  h16* wtD2  = (h16*)alloc(128 * 576 * 2);
  h16* wtR3a = (h16*)alloc(128 * 1152 * 2);
  h16* wtR3b = (h16*)alloc(128 * 1152 * 2);
  size_t slotBytes = (size_t)NP * 64 * 32 * 2;      // largest stage: 29.4 MB
  h16* slotA = (h16*)alloc(slotBytes);
  h16* slotB = (h16*)alloc(slotBytes);
  h16* slotC = (h16*)alloc(slotBytes);
  float* featF = (float*)alloc((size_t)NP * 128 * 4);
  h16*   featH = (h16*)alloc((size_t)NP * 128 * 2);
  int* topk = (int*)alloc(NB * TOPK * 4);
  int* srcE = (int*)alloc((size_t)NB * E_TOT * 4);
  int* dstE = (int*)alloc((size_t)NB * E_TOT * 4);
  int* mskE = (int*)alloc((size_t)NB * E_TOT * 4);
  h16* btWl1 = (h16*)alloc(128 * 128 * 2);
  h16* btWr1 = (h16*)alloc(128 * 128 * 2);
  h16* btWs1 = (h16*)alloc(128 * 128 * 2);
  h16* btWl2 = (h16*)alloc(128 * 256 * 2);
  h16* btWr2 = (h16*)alloc(128 * 256 * 2);
  h16* btWs2 = (h16*)alloc(128 * 256 * 2);
  float* xl1 = (float*)alloc((size_t)NP * 128 * 4);
  float* xr1 = (float*)alloc((size_t)NP * 128 * 4);
  float* out1 = (float*)alloc((size_t)NP * 128 * 4);
  h16*   h1H  = (h16*)alloc((size_t)NP * 128 * 2);
  float* xl2 = (float*)alloc((size_t)NP * 256 * 4);
  float* xr2 = (float*)alloc((size_t)NP * 256 * 4);
  float* out2 = (float*)alloc((size_t)NP * 256 * 4);
  float* esc = (float*)alloc((size_t)NB * E_TOT * 4);
  float* exv = (float*)alloc((size_t)NB * E_TOT * 4);
  unsigned* emax = (unsigned*)alloc(NP * 4);
  float* denom = (float*)alloc(NP * 4);
  float* gF = (float*)alloc((size_t)NB * TOPK * 256 * 4);
  h16*   gH = (h16*)alloc((size_t)NB * TOPK * 256 * 2);
  h16* btCw = (h16*)alloc(64 * 256 * 2);
  float* assignB = (float*)alloc((size_t)NB * TOPK * 64 * 4);
  float* asum = (float*)alloc(NB * 64 * 4);
  h16* btHw = (h16*)alloc((size_t)256 * 16384 * 2);
  float* vladF = (float*)alloc((size_t)NB * 16384 * 4);
  h16*   vladH = (h16*)alloc((size_t)NB * 16384 * 2);
  float* hidF = (float*)alloc(NB * 256 * 4);
  h16*   hidH = (h16*)alloc(NB * 256 * 2);
  h16* btGw = (h16*)alloc(256 * 256 * 2);
  float* gatesF = (float*)alloc(NB * 256 * 4);

  auto grid1 = [](long long n) { return dim3((unsigned)((n + 255) / 256)); };
  auto gemm = [&](const h16* A, const h16* Bt, const float* sc, const float* bi,
                  float* oF, h16* oH, int M, int N, int K, int actf) {
    int mt = (M + 15) / 16;
    dim3 g((mt + 3) / 4, N / 16);
    gemm_wmma<<<g, 128, 0, stream>>>(A, Bt, sc, bi, oF, oH, M, N, K, actf);
  };
  auto conv = [&](const h16* act, const h16* wt, const float* sc, const float* bi,
                  const h16* res, h16* outp, int IH, int IW, int Cin,
                  int OH, int OW, int Cout, int Kpad, int stride) {
    int M = NP * OH * OW;
    dim3 g((M / 16 + 3) / 4, Cout / 16);
    conv_wmma<<<g, 128, 0, stream>>>(act, wt, sc, bi, res, outp,
                                     NP, IH, IW, Cin, OH, OW, Cout, Kpad, stride, 1);
  };

  // --- weight packing ---
  pack_conv_w<<<grid1(32 * 32), 256, 0, stream>>>(c1w, wtC1, 32, 1, 32);
  pack_conv_w<<<grid1(32 * 288), 256, 0, stream>>>(r1w1, wtR1a, 32, 32, 288);
  pack_conv_w<<<grid1(32 * 288), 256, 0, stream>>>(r1w2, wtR1b, 32, 32, 288);
  pack_conv_w<<<grid1(64 * 288), 256, 0, stream>>>(d1w, wtD1, 64, 32, 288);
  pack_conv_w<<<grid1(64 * 576), 256, 0, stream>>>(r2w1, wtR2a, 64, 64, 576);
  pack_conv_w<<<grid1(64 * 576), 256, 0, stream>>>(r2w2, wtR2b, 64, 64, 576);
  pack_conv_w<<<grid1(128 * 576), 256, 0, stream>>>(d2w, wtD2, 128, 64, 576);
  pack_conv_w<<<grid1(128 * 1152), 256, 0, stream>>>(r3w1, wtR3a, 128, 128, 1152);
  pack_conv_w<<<grid1(128 * 1152), 256, 0, stream>>>(r3w2, wtR3b, 128, 128, 1152);
  pack_mat<<<grid1(128 * 128), 256, 0, stream>>>(g1wl, btWl1, 128, 128);
  pack_mat<<<grid1(128 * 128), 256, 0, stream>>>(g1wr, btWr1, 128, 128);
  pack_mat<<<grid1(128 * 128), 256, 0, stream>>>(g1wres, btWs1, 128, 128);
  pack_mat<<<grid1(128 * 256), 256, 0, stream>>>(g2wl, btWl2, 128, 256);
  pack_mat<<<grid1(128 * 256), 256, 0, stream>>>(g2wr, btWr2, 128, 256);
  pack_mat<<<grid1(128 * 256), 256, 0, stream>>>(g2wres, btWs2, 128, 256);
  pack_mat<<<grid1(256 * 64), 256, 0, stream>>>(vcw, btCw, 256, 64);
  pack_mat<<<grid1((long long)16384 * 256), 256, 0, stream>>>(vhw, btHw, 16384, 256);
  pack_mat<<<grid1(256 * 256), 256, 0, stream>>>(vgw, btGw, 256, 256);

  // --- pyramid CNN (WMMA implicit-GEMM, TDM-staged weights, fused epilogues) ---
  patchify<<<grid1(NP * 64), 256, 0, stream>>>(x, patchIn);
  conv(patchIn, wtC1, bn1g, bn1b, nullptr, slotA, 8, 8, 1, 8, 8, 32, 32, 1);
  conv(slotA, wtR1a, r1g1, r1b1, nullptr, slotB, 8, 8, 32, 8, 8, 32, 288, 1);
  conv(slotB, wtR1b, r1g2, r1b2, slotA, slotC, 8, 8, 32, 8, 8, 32, 288, 1);
  conv(slotC, wtD1, bnd1g, bnd1b, nullptr, slotA, 8, 8, 32, 4, 4, 64, 288, 2);
  conv(slotA, wtR2a, r2g1, r2b1, nullptr, slotB, 4, 4, 64, 4, 4, 64, 576, 1);
  conv(slotB, wtR2b, r2g2, r2b2, slotA, slotC, 4, 4, 64, 4, 4, 64, 576, 1);
  conv(slotC, wtD2, bnd2g, bnd2b, nullptr, slotA, 4, 4, 64, 2, 2, 128, 576, 2);
  conv(slotA, wtR3a, r3g1, r3b1, nullptr, slotB, 2, 2, 128, 2, 2, 128, 1152, 1);
  conv(slotB, wtR3b, r3g2, r3b2, slotA, slotC, 2, 2, 128, 2, 2, 128, 1152, 1);
  meanpool_feat<<<grid1(NP * 128), 256, 0, stream>>>(slotC, featF, featH);

  // --- top-K + edges ---
  topk_kernel<<<NB, 256, 0, stream>>>(featF, topk);
  build_edges<<<grid1((long long)NB * E_TOT), 256, 0, stream>>>(topk, srcE, dstE, mskE);

  // --- GAT layer 1 (dout=128) ---
  gemm(featH, btWl1, nullptr, g1bl, xl1, nullptr, NP, 128, 128, 0);
  gemm(featH, btWr1, nullptr, g1br, xr1, nullptr, NP, 128, 128, 0);
  gemm(featH, btWs1, nullptr, g1bias, out1, nullptr, NP, 128, 128, 0);
  init_seg<<<grid1(NP), 256, 0, stream>>>(emax, denom, NP);
  edge_score<<<grid1((long long)NB * E_TOT), 256, 0, stream>>>(xl1, xr1, g1att, srcE, dstE, mskE, esc, emax, 128);
  edge_exp<<<grid1((long long)NB * E_TOT), 256, 0, stream>>>(esc, dstE, mskE, emax, exv, denom);
  edge_aggr<<<4096, 256, 0, stream>>>(exv, denom, xl1, srcE, dstE, mskE, out1, 128);
  relu_store<<<grid1((long long)NP * 128), 256, 0, stream>>>(out1, h1H, (long long)NP * 128);

  // --- GAT layer 2 (dout=256) ---
  gemm(h1H, btWl2, nullptr, g2bl, xl2, nullptr, NP, 256, 128, 0);
  gemm(h1H, btWr2, nullptr, g2br, xr2, nullptr, NP, 256, 128, 0);
  gemm(h1H, btWs2, nullptr, g2bias, out2, nullptr, NP, 256, 128, 0);
  init_seg<<<grid1(NP), 256, 0, stream>>>(emax, denom, NP);
  edge_score<<<grid1((long long)NB * E_TOT), 256, 0, stream>>>(xl2, xr2, g2att, srcE, dstE, mskE, esc, emax, 256);
  edge_exp<<<grid1((long long)NB * E_TOT), 256, 0, stream>>>(esc, dstE, mskE, emax, exv, denom);
  edge_aggr<<<4096, 256, 0, stream>>>(exv, denom, xl2, srcE, dstE, mskE, out2, 256);
  relu_store<<<grid1((long long)NP * 256), 256, 0, stream>>>(out2, nullptr, (long long)NP * 256);

  // --- gather top-K nodes, L2 normalize -> g ---
  gather_norm<<<NB * TOPK, 256, 0, stream>>>(out2, topk, gF, gH);

  // --- NetVLAD ---
  gemm(gH, btCw, nullptr, vcb, assignB, nullptr, NB * TOPK, 64, 256, 0);
  softmax64<<<NB * TOPK, 64, 0, stream>>>(assignB);
  act_sum<<<grid1(NB * 64), 256, 0, stream>>>(assignB, asum);
  vlad_kernel<<<grid1((long long)NB * 16384), 256, 0, stream>>>(assignB, gF, vcw2, asum, vladF);
  vlad_intra_norm<<<NB * 64, 256, 0, stream>>>(vladF);
  vlad_global_norm<<<NB, 256, 0, stream>>>(vladF, vladH);
  gemm(vladH, btHw, vbn2g, vbn2b, hidF, hidH, NB, 256, 16384, 0);   // fused bn2
  gemm(hidH, btGw, nullptr, vgb, gatesF, nullptr, NB, 256, 256, 2); // sigmoid
  final_out<<<NB, 256, 0, stream>>>(hidF, gatesF, (float*)d_out);
}
